// _ECGNNLayer_12429635354688
// MI455X (gfx1250) — matrix operand order repaired and checked by reading, not compiled, not run on hardware
//
#include <hip/hip_runtime.h>

typedef _Float16 v16h __attribute__((ext_vector_type(16)));
typedef _Float16 v8h  __attribute__((ext_vector_type(8)));
typedef _Float16 v4h  __attribute__((ext_vector_type(4)));
typedef float    v8f  __attribute__((ext_vector_type(8)));

#define LN_EPS 1e-5f

// ---------- helpers ----------

__device__ __forceinline__ float hreduce16(float v) {
  // reduce across the 16 lanes of a half-wave (xor masks 1,2,4,8 stay within it)
  v += __shfl_xor(v, 1);
  v += __shfl_xor(v, 2);
  v += __shfl_xor(v, 4);
  v += __shfl_xor(v, 8);
  return v;
}

__device__ __forceinline__ float silu(float x) {
  return x / (1.0f + __expf(-x));
}

// Build a WMMA 16x32 f16 A fragment from row-major f16 LDS storage.
// For lane<16: halves 0..7 = K(base..base+7), halves 8..15 = K(base+16..base+23)
// For lane>=16 the caller passes base+8 (per ISA A-matrix layout).
__device__ __forceinline__ v16h load_afrag(const _Float16* p) {
  union { v16h v; v8h h[2]; } u;
  u.h[0] = *(const v8h*)(p);       // 16B ds_load_b128
  u.h[1] = *(const v8h*)(p + 16);  // 16B ds_load_b128
  return u.v;
}

// ---------- kernel 0: zero the aggregation buffer (== d_out) ----------

__global__ void zero_kernel(float* __restrict__ p, long long n) {
  long long i = (long long)blockIdx.x * blockDim.x + threadIdx.x;
  if (i < n) p[i] = 0.0f;
}

// ---------- kernel 1: pack weights into WMMA B-fragment layout ----------
// Fragment addressing: frag[((c*4 + n)*32 + lane)*16 + j]
//   lane<16 : column N = 16n + lane,    K = 32c + (j<8 ? j : 8 + j)
//   lane>=16: column N = 16n + lane-16, K = 32c + (j<8 ? 8 + j : 16 + j)
// (dense 16-bit B 32x16: lanes 0-15 hold K=0..7,16..23 of their column, lanes 16-31 K=8..15,24..31)

__device__ void pack_one(const float* __restrict__ W, int Kdim, _Float16* __restrict__ dst) {
  int total = Kdim * 64;
  for (int idx = threadIdx.x; idx < total; idx += blockDim.x) {
    int j    = idx & 15;
    int lane = (idx >> 4) & 31;
    int rest = idx >> 9;
    int n    = rest & 3;
    int c    = rest >> 2;
    int col  = 16 * n + (lane & 15);
    int k    = 32 * c + 8 * (lane >> 4) + ((j < 8) ? j : 8 + j);
    dst[idx] = (_Float16)W[k * 64 + col];
  }
}

__global__ void pack_weights_kernel(const float* __restrict__ W1,
                                    const float* __restrict__ W2,
                                    const float* __restrict__ Wu,
                                    _Float16* __restrict__ ws) {
  pack_one(W1, 96,  ws);
  pack_one(W2, 64,  ws + 96 * 64);
  pack_one(Wu, 128, ws + 96 * 64 + 64 * 64);
}

// ---------- kernel 2: edge MLP + scatter ----------
// block = 256 threads (8 waves), 128 edges/block, 16 edges/wave.

__global__ __launch_bounds__(256, 2)
void edge_kernel(const float* __restrict__ nodef,
                 const int*   __restrict__ eidx,     // [2, E]
                 const float* __restrict__ edgef,    // [E, 32]
                 const float* __restrict__ b1,
                 const float* __restrict__ g1,
                 const float* __restrict__ be1,
                 const float* __restrict__ b2,
                 const _Float16* __restrict__ B1,
                 const _Float16* __restrict__ B2,
                 float* __restrict__ agg,            // [N, 64] (atomic accumulate)
                 long long E) {
  __shared__ __align__(16) _Float16 Xs[128][104];   // 128 x 96 f16, padded stride
  __shared__ __align__(16) _Float16 Hs[8][16][72];  // per-wave h staging
  __shared__ int srcs[128];
  __shared__ int tgts[128];

  const int tid = threadIdx.x;
  const long long e0 = (long long)blockIdx.x * 128;

  if (tid < 128) {
    long long e = e0 + tid;
    if (e >= E) e = E - 1;
    srcs[tid] = eidx[e];
    tgts[tid] = eidx[E + e];
  }
  __syncthreads();

  // stage X = [gathered src feats (64) | edge feats (32)] as f16
  for (int i = tid; i < 128 * 24; i += 256) {
    int r   = i / 24;
    int col = (i % 24) * 4;
    long long er = e0 + r; if (er >= E) er = E - 1;
    float4 v;
    if (col < 64) v = *(const float4*)(nodef + (size_t)srcs[r] * 64 + col);
    else          v = *(const float4*)(edgef + er * 32 + (col - 64));
    v4h h;
    h[0] = (_Float16)v.x; h[1] = (_Float16)v.y;
    h[2] = (_Float16)v.z; h[3] = (_Float16)v.w;
    *(v4h*)&Xs[r][col] = h;
  }
  __syncthreads();

  const int wave  = tid >> 5;
  const int lane  = tid & 31;
  const int lan16 = lane & 15;
  const int hi    = lane >> 4;
  const int r0    = wave * 16;

  // ---- layer 1: (16x96) @ (96x64), f32 accumulate ----
  v8f acc[4];
#pragma unroll
  for (int n = 0; n < 4; ++n)
#pragma unroll
    for (int j = 0; j < 8; ++j) acc[n][j] = 0.0f;

#pragma unroll
  for (int c = 0; c < 3; ++c) {
    v16h a = load_afrag(&Xs[r0 + lan16][32 * c + 8 * hi]);
#pragma unroll
    for (int n = 0; n < 4; ++n) {
      v16h b = *(const v16h*)(B1 + (((c * 4) + n) * 32 + lane) * 16);
      acc[n] = __builtin_amdgcn_wmma_f32_16x16x32_f16(
          false, a, false, b, (short)0, acc[n], false, false);
    }
  }

  // bias + LayerNorm(64) + SiLU, write h back to LDS in f16
  float gv[4], bev[4];
#pragma unroll
  for (int n = 0; n < 4; ++n) {
    int f = 16 * n + lan16;
    float bb = b1[f];
    gv[n] = g1[f]; bev[n] = be1[f];
#pragma unroll
    for (int j = 0; j < 8; ++j) acc[n][j] += bb;
  }
#pragma unroll
  for (int j = 0; j < 8; ++j) {
    float s = 0.0f, q = 0.0f;
#pragma unroll
    for (int n = 0; n < 4; ++n) { float x = acc[n][j]; s += x; q += x * x; }
    s = hreduce16(s); q = hreduce16(q);
    float mu = s * (1.0f / 64.0f);
    float rs = rsqrtf(q * (1.0f / 64.0f) - mu * mu + LN_EPS);
    int m = j + 8 * hi;
#pragma unroll
    for (int n = 0; n < 4; ++n) {
      float x = (acc[n][j] - mu) * rs * gv[n] + bev[n];
      Hs[wave][m][16 * n + lan16] = (_Float16)silu(x);
    }
  }
  __syncthreads();

  // ---- layer 2: (16x64) @ (64x64) ----
  v8f acc2[4];
#pragma unroll
  for (int n = 0; n < 4; ++n)
#pragma unroll
    for (int j = 0; j < 8; ++j) acc2[n][j] = 0.0f;

#pragma unroll
  for (int c = 0; c < 2; ++c) {
    v16h a = load_afrag(&Hs[wave][lan16][32 * c + 8 * hi]);
#pragma unroll
    for (int n = 0; n < 4; ++n) {
      v16h b = *(const v16h*)(B2 + (((c * 4) + n) * 32 + lane) * 16);
      acc2[n] = __builtin_amdgcn_wmma_f32_16x16x32_f16(
          false, a, false, b, (short)0, acc2[n], false, false);
    }
  }

  // bias + SiLU + scatter-add messages into target nodes (L2-resident)
  float b2v[4];
#pragma unroll
  for (int n = 0; n < 4; ++n) b2v[n] = b2[16 * n + lan16];
#pragma unroll
  for (int j = 0; j < 8; ++j) {
    int m = j + 8 * hi;
    float* dst = agg + (size_t)tgts[r0 + m] * 64;
#pragma unroll
    for (int n = 0; n < 4; ++n) {
      float x = silu(acc2[n][j] + b2v[n]);
      unsafeAtomicAdd(dst + 16 * n + lan16, x);  // global_atomic_add_f32
    }
  }
}

// ---------- kernel 3: node update ----------
// combined = [agg (64) | node_features (64)], (16x128)@(128x64), SiLU, LayerNorm.

__global__ __launch_bounds__(256, 2)
void node_kernel(const float* __restrict__ nodef,
                 const float* __restrict__ bu,
                 const float* __restrict__ gn,
                 const float* __restrict__ bn,
                 const _Float16* __restrict__ BU,
                 float* __restrict__ out,   // in: agg rows, out: final rows
                 long long N) {
  __shared__ __align__(16) _Float16 Cs[128][136];  // 128 x 128 f16, padded stride

  const int tid = threadIdx.x;
  const long long n0 = (long long)blockIdx.x * 128;

  for (int i = tid; i < 128 * 32; i += 256) {
    int r   = i / 32;
    int col = (i % 32) * 4;
    long long nr = n0 + r; if (nr >= N) nr = N - 1;
    float4 v;
    if (col < 64) v = *(const float4*)(out   + nr * 64 + col);
    else          v = *(const float4*)(nodef + nr * 64 + (col - 64));
    v4h h;
    h[0] = (_Float16)v.x; h[1] = (_Float16)v.y;
    h[2] = (_Float16)v.z; h[3] = (_Float16)v.w;
    *(v4h*)&Cs[r][col] = h;
  }
  __syncthreads();

  const int wave  = tid >> 5;
  const int lane  = tid & 31;
  const int lan16 = lane & 15;
  const int hi    = lane >> 4;
  const int r0    = wave * 16;

  v8f acc[4];
#pragma unroll
  for (int n = 0; n < 4; ++n)
#pragma unroll
    for (int j = 0; j < 8; ++j) acc[n][j] = 0.0f;

#pragma unroll
  for (int c = 0; c < 4; ++c) {
    v16h a = load_afrag(&Cs[r0 + lan16][32 * c + 8 * hi]);
#pragma unroll
    for (int n = 0; n < 4; ++n) {
      v16h b = *(const v16h*)(BU + (((c * 4) + n) * 32 + lane) * 16);
      acc[n] = __builtin_amdgcn_wmma_f32_16x16x32_f16(
          false, a, false, b, (short)0, acc[n], false, false);
    }
  }

  float buv[4], gnv[4], bnv[4];
#pragma unroll
  for (int n = 0; n < 4; ++n) {
    int f = 16 * n + lan16;
    buv[n] = bu[f]; gnv[n] = gn[f]; bnv[n] = bn[f];
  }
#pragma unroll
  for (int j = 0; j < 8; ++j) {
    float x4[4];
    float s = 0.0f, q = 0.0f;
#pragma unroll
    for (int n = 0; n < 4; ++n) {
      float x = silu(acc[n][j] + buv[n]);
      x4[n] = x; s += x; q += x * x;
    }
    s = hreduce16(s); q = hreduce16(q);
    float mu = s * (1.0f / 64.0f);
    float rs = rsqrtf(q * (1.0f / 64.0f) - mu * mu + LN_EPS);
    int m = j + 8 * hi;
    long long row = n0 + r0 + m; if (row >= N) row = N - 1;
    float* dst = out + row * 64;
#pragma unroll
    for (int n = 0; n < 4; ++n)
      dst[16 * n + lan16] = (x4[n] - mu) * rs * gnv[n] + bnv[n];
  }
}

// ---------- launch ----------

extern "C" void kernel_launch(void* const* d_in, const int* in_sizes, int n_in,
                              void* d_out, int out_size, void* d_ws, size_t ws_size,
                              hipStream_t stream) {
  const float* nodef = (const float*)d_in[0];
  const int*   eidx  = (const int*)d_in[1];
  const float* edgef = (const float*)d_in[2];
  const float* W1    = (const float*)d_in[3];
  const float* b1    = (const float*)d_in[4];
  const float* g1    = (const float*)d_in[5];
  const float* be1   = (const float*)d_in[6];
  const float* W2    = (const float*)d_in[7];
  const float* b2    = (const float*)d_in[8];
  const float* Wu    = (const float*)d_in[9];
  const float* bu    = (const float*)d_in[10];
  const float* gn    = (const float*)d_in[11];
  const float* bn    = (const float*)d_in[12];

  const long long N = in_sizes[0] / 64;   // node count
  const long long E = in_sizes[2] / 32;   // edge count

  float*    out = (float*)d_out;
  _Float16* ws  = (_Float16*)d_ws;        // packed weights: 36 KB total
  _Float16* B1  = ws;
  _Float16* B2  = ws + 96 * 64;
  _Float16* BU  = ws + 96 * 64 + 64 * 64;

  const long long outElems = N * 64;
  int zblocks = (int)((outElems + 255) / 256);
  zero_kernel<<<zblocks, 256, 0, stream>>>(out, outElems);

  pack_weights_kernel<<<1, 256, 0, stream>>>(W1, W2, Wu, ws);

  int eblocks = (int)((E + 127) / 128);
  edge_kernel<<<eblocks, 256, 0, stream>>>(nodef, eidx, edgef,
                                           b1, g1, be1, b2, B1, B2, out, E);

  int nblocks = (int)((N + 127) / 128);
  node_kernel<<<nblocks, 256, 0, stream>>>(nodef, bu, gn, bn, BU, out, N);
}
